// SpatioTemporalAttention_12867722019533
// MI455X (gfx1250) — compile-verified, matrix operand order checked by reading, not run on hardware
//
#include <hip/hip_runtime.h>
#include <hip/hip_bf16.h>

// ---------------------------------------------------------------------------
// SpatioTemporalAttention for MI455X (gfx1250, wave32, WMMA + async-LDS).
//
// Roofline: 4x (32768x512)x(512x512) GEMMs ~= 69 GFLOP; min HBM traffic
// ~201 MB -> ~8.6 us @ 23.3 TB/s.  bf16 WMMA makes the GEMMs memory-bound,
// so the GEMM inner loop is pure: async global->LDS staging (ASYNCcnt,
// double-buffered) feeding v_wmma_f32_16x16x32_bf16.  Activations are
// converted to bf16 once up front; intermediates live in workspace.
// ---------------------------------------------------------------------------

typedef __attribute__((ext_vector_type(16))) __bf16         v16bf;
typedef __attribute__((ext_vector_type(16))) unsigned short v16u;
typedef __attribute__((ext_vector_type(8)))  unsigned short v8u;
typedef __attribute__((ext_vector_type(8)))  float          v8f;
typedef __attribute__((ext_vector_type(4)))  float          v4f;
typedef __attribute__((ext_vector_type(4)))  int            v4i;

union FragU { v16u u; v16bf b; };

__device__ __forceinline__ unsigned short f2bf(float f) {
    unsigned int u = __float_as_uint(f);
    u += 0x7FFFu + ((u >> 16) & 1u);   // round-to-nearest-even
    return (unsigned short)(u >> 16);
}

__device__ __forceinline__ v8f v8f_zero() {
    v8f z = {0.f, 0.f, 0.f, 0.f, 0.f, 0.f, 0.f, 0.f};
    return z;
}

// --- async global->LDS 16B copy + ASYNCcnt wait --------------------------
#if defined(__HIP_DEVICE_COMPILE__) && __has_builtin(__builtin_amdgcn_global_load_async_to_lds_b128)
__device__ __forceinline__ void cp_async_b128(const unsigned short* g, unsigned short* l) {
    __builtin_amdgcn_global_load_async_to_lds_b128(
        (__attribute__((address_space(1))) v4i*)g,
        (__attribute__((address_space(3))) v4i*)l, 0, 0);
}
#else
__device__ __forceinline__ void cp_async_b128(const unsigned short* g, unsigned short* l) {
    unsigned lofs = (unsigned)(size_t)l;   // LDS aperture: low 32 bits = LDS offset
    asm volatile("global_load_async_to_lds_b128 %0, %1, off"
                 :: "v"(lofs), "v"(g) : "memory");
}
#endif

#if defined(__HIP_DEVICE_COMPILE__) && __has_builtin(__builtin_amdgcn_s_wait_asynccnt)
#define WAIT_ASYNC(n) __builtin_amdgcn_s_wait_asynccnt(n)
#else
#define WAIT_ASYNC(n) asm volatile("s_wait_asynccnt %0" :: "i"(n) : "memory")
#endif

// ---------------------------------------------------------------------------
// fp32 -> bf16 (as ushort), 8 elements per thread
// ---------------------------------------------------------------------------
__global__ void cvt_f32_bf16(const float* __restrict__ src,
                             unsigned short* __restrict__ dst, int n8) {
    int i = blockIdx.x * blockDim.x + threadIdx.x;
    if (i >= n8) return;
    const float* s = src + (size_t)i * 8;
    v4f a = *(const v4f*)s;
    v4f b = *(const v4f*)(s + 4);
    union { v8u v; unsigned short s[8]; } o;
    o.s[0] = f2bf(a[0]); o.s[1] = f2bf(a[1]); o.s[2] = f2bf(a[2]); o.s[3] = f2bf(a[3]);
    o.s[4] = f2bf(b[0]); o.s[5] = f2bf(b[1]); o.s[6] = f2bf(b[2]); o.s[7] = f2bf(b[3]);
    *(v8u*)(dst + (size_t)i * 8) = o.v;
}

// ---------------------------------------------------------------------------
// Y[M,512] = A[M,512](bf16) @ W[512,512](bf16)^T + bias  -> fp32
// 128x128 block tile, K-step 32, double-buffered async global->LDS staging.
// 8 waves; wave tile 32x64 = 2x4 WMMA tiles of 16x16x32 bf16.
// ---------------------------------------------------------------------------
__global__ __launch_bounds__(256) void gemm_nt_bias(
    const unsigned short* __restrict__ A, const unsigned short* __restrict__ W,
    const float* __restrict__ bias, float* __restrict__ Y)
{
    __shared__ alignas(32) unsigned short As[2][128 * 48];  // row stride 48 (96B)
    __shared__ alignas(32) unsigned short Bs[2][128 * 48];

    const int tid  = threadIdx.x;
    const int m0   = blockIdx.x * 128;
    const int n0   = blockIdx.y * 128;
    const int wave = tid >> 5;
    const int lane = tid & 31;
    const int wm   = (wave & 3) * 32;   // wave M offset inside block tile
    const int wn   = (wave >> 2) * 64;  // wave N offset inside block tile

    const int lrow  = tid >> 1;         // 0..127 : tile row this thread stages
    const int lhalf = (tid & 1) * 16;   // 0 or 16 : K sub-chunk

    const unsigned short* gA = A + (size_t)(m0 + lrow) * 512 + lhalf;
    const unsigned short* gB = W + (size_t)(n0 + lrow) * 512 + lhalf;
    const int lofs = lrow * 48 + lhalf;

    v8f acc[2][4];
#pragma unroll
    for (int mt = 0; mt < 2; ++mt)
#pragma unroll
        for (int nt = 0; nt < 4; ++nt) acc[mt][nt] = v8f_zero();

    // prologue: stage K-chunk 0 into buffer 0 (4 async b128 per thread)
    cp_async_b128(gA,     &As[0][lofs]);
    cp_async_b128(gA + 8, &As[0][lofs + 8]);
    cp_async_b128(gB,     &Bs[0][lofs]);
    cp_async_b128(gB + 8, &Bs[0][lofs + 8]);

    const int fr = lane & 15;           // row/col within 16
    const int fk = (lane >> 4) * 16;    // K sub-chunk per half-wave

    for (int k0 = 0, it = 0; k0 < 512; k0 += 32, ++it) {
        const int cur = it & 1;
        if (k0 + 32 < 512) {
            // stage next K-chunk into the other buffer, then wait only for
            // the current chunk (per-wave async loads complete in order).
            const int nb = cur ^ 1;
            const unsigned short* nA = gA + k0 + 32;
            const unsigned short* nB = gB + k0 + 32;
            cp_async_b128(nA,     &As[nb][lofs]);
            cp_async_b128(nA + 8, &As[nb][lofs + 8]);
            cp_async_b128(nB,     &Bs[nb][lofs]);
            cp_async_b128(nB + 8, &Bs[nb][lofs + 8]);
            WAIT_ASYNC(4);
        } else {
            WAIT_ASYNC(0);
        }
        __syncthreads();

#pragma unroll
        for (int mt = 0; mt < 2; ++mt) {
            FragU fa;
            fa.u = *(const v16u*)(&As[cur][(wm + mt * 16 + fr) * 48 + fk]);
#pragma unroll
            for (int nt = 0; nt < 4; ++nt) {
                FragU fb;
                fb.u = *(const v16u*)(&Bs[cur][(wn + nt * 16 + fr) * 48 + fk]);
                acc[mt][nt] = __builtin_amdgcn_wmma_f32_16x16x32_bf16(
                    false, fa.b, false, fb.b, (short)0, acc[mt][nt], false, false);
            }
        }
        __syncthreads();   // protect buffer reuse before next issue
    }

    // epilogue: D layout (lanes0-15: M=j, lanes16-31: M=j+8; N=lane&15)
    const int crow = (lane >> 4) * 8;
    const int ccol = lane & 15;
#pragma unroll
    for (int mt = 0; mt < 2; ++mt) {
#pragma unroll
        for (int nt = 0; nt < 4; ++nt) {
            const int   oc = n0 + wn + nt * 16 + ccol;
            const float bv = bias[oc];
            float* yp = Y + (size_t)(m0 + wm + mt * 16 + crow) * 512 + oc;
#pragma unroll
            for (int j = 0; j < 8; ++j)
                yp[(size_t)j * 512] = acc[mt][nt][j] + bv;
        }
    }
}

// ---------------------------------------------------------------------------
// Per-(n,h) additive pooling with softmax over L=2048.
//   score_l = (sum_d X[n,l,h*64+d] * gv[d] + b) / 8,  gv = wvec (* gate)
//   pooled[n,h,d] = sum_l softmax(score)_l * X[n,l,h*64+d]
// ---------------------------------------------------------------------------
__global__ __launch_bounds__(256) void pool_kernel(
    const float* __restrict__ X, const float* __restrict__ wvec,
    const float* __restrict__ bsc, const float* __restrict__ gate,
    float* __restrict__ pooled)
{
    __shared__ float sc[2048];
    __shared__ float gv[64];
    __shared__ float red[256];

    const int n   = blockIdx.x >> 3;
    const int h   = blockIdx.x & 7;
    const int tid = threadIdx.x;

    if (tid < 64) {
        float g = gate ? gate[((size_t)n * 8 + h) * 64 + tid] : 1.f;
        gv[tid] = wvec[tid] * g;
    }
    __syncthreads();

    const float* base = X + (size_t)n * 2048 * 512 + h * 64;
    const float  b0   = bsc[0];

    for (int l = tid; l < 2048; l += 256) {
        const float* xp = base + (size_t)l * 512;
        float s = 0.f;
#pragma unroll
        for (int d = 0; d < 64; d += 4) {
            v4f t = *(const v4f*)(xp + d);
            s += t[0] * gv[d] + t[1] * gv[d + 1] + t[2] * gv[d + 2] + t[3] * gv[d + 3];
        }
        sc[l] = (s + b0) * 0.125f;   // 1/sqrt(64)
    }
    __syncthreads();

    float m = -3.4e38f;
    for (int l = tid; l < 2048; l += 256) m = fmaxf(m, sc[l]);
    red[tid] = m;
    __syncthreads();
    for (int s = 128; s > 0; s >>= 1) {
        if (tid < s) red[tid] = fmaxf(red[tid], red[tid + s]);
        __syncthreads();
    }
    const float mx = red[0];
    __syncthreads();

    float sum = 0.f;
    for (int l = tid; l < 2048; l += 256) {
        float e = __expf(sc[l] - mx);
        sc[l] = e;
        sum += e;
    }
    red[tid] = sum;
    __syncthreads();
    for (int s = 128; s > 0; s >>= 1) {
        if (tid < s) red[tid] += red[tid + s];
        __syncthreads();
    }
    const float rinv = 1.f / red[0];
    __syncthreads();

    const int d  = tid & 63;
    const int lg = tid >> 6;
    float acc = 0.f;
    for (int l = lg; l < 2048; l += 4) acc += sc[l] * base[(size_t)l * 512 + d];
    red[tid] = acc;
    __syncthreads();
    if (tid < 64)
        pooled[((size_t)n * 8 + h) * 64 + tid] =
            (red[tid] + red[tid + 64] + red[tid + 128] + red[tid + 192]) * rinv;
}

// ---------------------------------------------------------------------------
// out_pre[n,l,h*64+e] = sum_d (v[n,l,h*64+d]*pooled_k[n,h,d]) * Wt[e,d]
//                       + bt[e] + q[n,l,h*64+e]        (written as bf16)
// Per-head 64x64 transform via WMMA (K=64 -> two bf16 WMMAs per tile).
// Grid: (M/128, H); 8 waves, each wave a 16-row strip x 64 cols (4 tiles).
// ---------------------------------------------------------------------------
__global__ __launch_bounds__(256) void head_mix_kernel(
    const float* __restrict__ v, const float* __restrict__ q,
    const unsigned short* __restrict__ Wtb, const float* __restrict__ bt,
    const float* __restrict__ pooled_k, unsigned short* __restrict__ outp)
{
    __shared__ alignas(32) unsigned short WtS[64 * 80];  // row stride 80 (160B)
    __shared__ float gk[64];
    __shared__ float bts[64];

    const int tid  = threadIdx.x;
    const int h    = blockIdx.y;
    const int n    = blockIdx.x >> 4;       // 128 rows per block, L=2048
    const int lane = tid & 31;

    {
        const int r = tid >> 2, seg = (tid & 3) * 16;
        *(v16u*)(WtS + r * 80 + seg) = *(const v16u*)(Wtb + r * 64 + seg);
    }
    if (tid < 64) {
        gk[tid]  = pooled_k[((size_t)n * 8 + h) * 64 + tid];
        bts[tid] = bt[tid];
    }
    __syncthreads();

    const int rbase = blockIdx.x * 128 + (tid >> 5) * 16;
    const int fr = lane & 15;
    const int fk = (lane >> 4) * 16;

    v8f acc[4];
#pragma unroll
    for (int nt = 0; nt < 4; ++nt) acc[nt] = v8f_zero();

#pragma unroll
    for (int kc = 0; kc < 2; ++kc) {
        const int dbase = kc * 32 + fk;
        const float* vp = v + (size_t)(rbase + fr) * 512 + h * 64 + dbase;
        FragU fa;
#pragma unroll
        for (int j = 0; j < 16; j += 4) {
            v4f t = *(const v4f*)(vp + j);
            fa.u[j + 0] = f2bf(t[0] * gk[dbase + j + 0]);
            fa.u[j + 1] = f2bf(t[1] * gk[dbase + j + 1]);
            fa.u[j + 2] = f2bf(t[2] * gk[dbase + j + 2]);
            fa.u[j + 3] = f2bf(t[3] * gk[dbase + j + 3]);
        }
#pragma unroll
        for (int nt = 0; nt < 4; ++nt) {
            FragU fb;
            fb.u = *(const v16u*)(WtS + (nt * 16 + fr) * 80 + dbase);
            acc[nt] = __builtin_amdgcn_wmma_f32_16x16x32_bf16(
                false, fa.b, false, fb.b, (short)0, acc[nt], false, false);
        }
    }

    const int crow = (lane >> 4) * 8;
#pragma unroll
    for (int nt = 0; nt < 4; ++nt) {
        const int   e  = nt * 16 + (lane & 15);
        const float be = bts[e];
#pragma unroll
        for (int j = 0; j < 8; ++j) {
            const size_t idx = (size_t)(rbase + crow + j) * 512 + h * 64 + e;
            outp[idx] = f2bf(acc[nt][j] + be + q[idx]);
        }
    }
}

// ---------------------------------------------------------------------------
// Launcher
// ---------------------------------------------------------------------------
extern "C" void kernel_launch(void* const* d_in, const int* in_sizes, int n_in,
                              void* d_out, int out_size, void* d_ws, size_t ws_size,
                              hipStream_t stream) {
    const float* x_q  = (const float*)d_in[0];
    const float* x_kv = (const float*)d_in[1];
    const float* Wq   = (const float*)d_in[2];
    const float* bq   = (const float*)d_in[3];
    const float* wqa  = (const float*)d_in[4];
    const float* bqa  = (const float*)d_in[5];
    const float* Wk   = (const float*)d_in[6];
    const float* bk   = (const float*)d_in[7];
    const float* wka  = (const float*)d_in[8];
    const float* bka  = (const float*)d_in[9];
    const float* Wv   = (const float*)d_in[10];
    const float* bv   = (const float*)d_in[11];
    const float* Wt   = (const float*)d_in[12];
    const float* bt   = (const float*)d_in[13];
    const float* Wp   = (const float*)d_in[14];
    const float* bp   = (const float*)d_in[15];
    float* out = (float*)d_out;

    const int M  = in_sizes[0] / 512;   // N*L = 32768
    const int NB = M / 2048;            // N = 16

    // workspace layout
    float* qbuf = (float*)d_ws;                        // M*512 f32
    float* kbuf = qbuf + (size_t)M * 512;              // M*512 f32
    float* vbuf = kbuf + (size_t)M * 512;              // M*512 f32
    unsigned short* xq_bf  = (unsigned short*)(vbuf + (size_t)M * 512);  // M*512 bf16 (reused as out_pre)
    unsigned short* xkv_bf = xq_bf + (size_t)M * 512;                    // M*512 bf16
    unsigned short* Wqb = xkv_bf + (size_t)M * 512;
    unsigned short* Wkb = Wqb + 512 * 512;
    unsigned short* Wvb = Wkb + 512 * 512;
    unsigned short* Wpb = Wvb + 512 * 512;
    unsigned short* Wtb = Wpb + 512 * 512;
    float* pooled_q = (float*)(Wtb + 64 * 64);
    float* pooled_k = pooled_q + (size_t)NB * 8 * 64;

    // 1) fp32 -> bf16: activations and weights
    const int n8x = (M * 512) / 8;
    cvt_f32_bf16<<<(n8x + 255) / 256, 256, 0, stream>>>(x_q,  xq_bf,  n8x);
    cvt_f32_bf16<<<(n8x + 255) / 256, 256, 0, stream>>>(x_kv, xkv_bf, n8x);
    cvt_f32_bf16<<<128, 256, 0, stream>>>(Wq, Wqb, 512 * 512 / 8);
    cvt_f32_bf16<<<128, 256, 0, stream>>>(Wk, Wkb, 512 * 512 / 8);
    cvt_f32_bf16<<<128, 256, 0, stream>>>(Wv, Wvb, 512 * 512 / 8);
    cvt_f32_bf16<<<128, 256, 0, stream>>>(Wp, Wpb, 512 * 512 / 8);
    cvt_f32_bf16<<<2,   256, 0, stream>>>(Wt, Wtb, 64 * 64 / 8);

    dim3 ggrid(M / 128, 4);

    // 2) q = x_q @ Wq^T + bq
    gemm_nt_bias<<<ggrid, 256, 0, stream>>>(xq_bf, Wqb, bq, qbuf);
    // 3) pooled_q
    pool_kernel<<<NB * 8, 256, 0, stream>>>(qbuf, wqa, bqa, nullptr, pooled_q);
    // 4) k = x_kv @ Wk^T + bk
    gemm_nt_bias<<<ggrid, 256, 0, stream>>>(xkv_bf, Wkb, bk, kbuf);
    // 5) pooled_k (scores gated by pooled_q)
    pool_kernel<<<NB * 8, 256, 0, stream>>>(kbuf, wka, bka, pooled_q, pooled_k);
    // 6) v = x_kv @ Wv^T + bv
    gemm_nt_bias<<<ggrid, 256, 0, stream>>>(xkv_bf, Wvb, bv, vbuf);
    // 7) out_pre(bf16) = (v*pooled_k)@Wt^T + bt + q   (overwrites dead xq_bf)
    dim3 hgrid(M / 128, 8);
    head_mix_kernel<<<hgrid, 256, 0, stream>>>(vbuf, qbuf, Wtb, bt, pooled_k, xq_bf);
    // 8) out = out_pre @ Wp^T + bp
    gemm_nt_bias<<<ggrid, 256, 0, stream>>>(xq_bf, Wpb, bp, out);
}
